// LDM_TRI_1245540516213
// MI455X (gfx1250) — matrix operand order repaired and checked by reading, not compiled, not run on hardware
//
#include <hip/hip_runtime.h>
#include <hip/hip_bf16.h>
#include <math.h>

#define EPS_F 1e-6f

typedef __attribute__((ext_vector_type(16))) __bf16 v16bf;
typedef __attribute__((ext_vector_type(8)))  __bf16 bf16x8;
typedef __attribute__((ext_vector_type(8)))  float  v8f;
typedef __attribute__((ext_vector_type(4)))  float  v4f;

// ---------------------------------------------------------------------------
// A-fragment (16x32 bf16, MxK): lane l holds row (l&15); K-chunks
// [ (l>>4)*8 , +8 ) and [ 16+(l>>4)*8 , +8 )  -> two contiguous 16B loads.
// ---------------------------------------------------------------------------
static __device__ __forceinline__ v16bf load_a_frag(const __bf16* __restrict__ p,
                                                    int row0, int kb, int m, int hl) {
  const __bf16* q = p + (size_t)(row0 + m) * 128 + kb + hl * 8;
  bf16x8 c0 = *(const bf16x8*)q;
  bf16x8 c1 = *(const bf16x8*)(q + 16);
  v16bf r;
#pragma unroll
  for (int t = 0; t < 8; ++t) { r[t] = c0[t]; r[t + 8] = c1[t]; }
  return r;
}

// ---------------------------------------------------------------------------
// B-fragment (32x16 bf16, KxN), B = L^T with L row-major: lane l holds column
// (l&15) = row of L, K-chunks [ (l>>4)*8 , +8 ) and [ 16+(l>>4)*8 , +8 ).
// (These are J-loop invariant; the compiler hoists them into high VGPRs.)
// ---------------------------------------------------------------------------
static __device__ __forceinline__ v16bf load_b_frag(const __bf16* __restrict__ p,
                                                    int col0, int kb, int n, int hl) {
  const __bf16* q = p + (size_t)(col0 + n) * 128 + kb + hl * 8;
  bf16x8 c0 = *(const bf16x8*)q;
  bf16x8 c1 = *(const bf16x8*)(q + 16);
  v16bf r;
#pragma unroll
  for (int t = 0; t < 8; ++t) { r[t] = c0[t]; r[t + 8] = c1[t]; }
  return r;
}

// ---------------------------------------------------------------------------
// fp32 -> bf16 hi/lo split + row squared-norms.  One wave per row (D=128,
// 4 elements per lane).
// ---------------------------------------------------------------------------
__global__ __launch_bounds__(256) void ldm_prep_kernel(
    const float* __restrict__ src, int rows, float eps,
    __bf16* __restrict__ hi, __bf16* __restrict__ lo, float* __restrict__ nrm) {
  const int wave = threadIdx.x >> 5;
  const int lane = threadIdx.x & 31;
  const int row  = blockIdx.x * 8 + wave;
  if (row >= rows) return;
  const float* s = src + (size_t)row * 128 + lane * 4;
  __bf16* h = hi + (size_t)row * 128 + lane * 4;
  __bf16* l = lo + (size_t)row * 128 + lane * 4;
  float acc = 0.f;
#pragma unroll
  for (int d = 0; d < 4; ++d) {
    float x   = s[d] + eps;
    __bf16 xh = (__bf16)x;
    h[d] = xh;
    l[d] = (__bf16)(x - (float)xh);
    acc += x * x;
  }
#pragma unroll
  for (int off = 16; off > 0; off >>= 1) acc += __shfl_xor(acc, off, 32);
  if (lane == 0) nrm[row] = acc;
}

// ---------------------------------------------------------------------------
// Fused "rate" pass:
//   colPart[blockIdx.y][i] = sum_{j in partition} exp( abias[j] -
//       (sqrt(max(anorm[j] + bnorm[i] - 2*(A·B^T)[j,i], 0)) + EPS) )
// A (rows x 128) via bf16 hi/lo split WMMA against B = latent_l (cols x 128).
// Block: 64 i-columns x 128 j-rows per iteration; 8 waves, wave w owns
// j-strip [w*16, w*16+16) x 64 columns = 4 WMMA accumulators.
// Fully deterministic: no atomics — per-wave LDS stores + fixed-order reduce,
// per-partition output slices combined in the finalize kernel.
// ---------------------------------------------------------------------------
__global__ __launch_bounds__(256) void ldm_rate_colsum_kernel(
    const __bf16* __restrict__ Ahi, const __bf16* __restrict__ Alo,
    const float*  __restrict__ anorm, const float* __restrict__ abias,
    const __bf16* __restrict__ Bhi, const __bf16* __restrict__ Blo,
    const float*  __restrict__ bnorm,
    int rows_per_part, int colStride, float* __restrict__ colPart) {
  const int ic0  = blockIdx.x * 64;
  const int jp0  = blockIdx.y * rows_per_part;
  const int wave = threadIdx.x >> 5;
  const int lane = threadIdx.x & 31;
  const int m    = lane & 15;   // row-in-tile (A) / column-in-tile (B,C)
  const int hl   = lane >> 4;   // lane half selects K sub-range / M+8 rows

  __shared__ float lds_part[8][64];
  float colacc = 0.f;           // live in threads 0..63

  float nli[4];
#pragma unroll
  for (int t = 0; t < 4; ++t) nli[t] = bnorm[ic0 + t * 16 + m];

  for (int jr = jp0; jr < jp0 + rows_per_part; jr += 128) {
    const int jbase = jr + wave * 16;

    v8f acc[4];
#pragma unroll
    for (int t = 0; t < 4; ++t)
#pragma unroll
      for (int p = 0; p < 8; ++p) acc[t][p] = 0.f;

#pragma unroll
    for (int kb = 0; kb < 128; kb += 32) {
      const v16bf ahi = load_a_frag(Ahi, jbase, kb, m, hl);
      const v16bf alo = load_a_frag(Alo, jbase, kb, m, hl);
#pragma unroll
      for (int t = 0; t < 4; ++t) {
        const v16bf bhi = load_b_frag(Bhi, ic0 + t * 16, kb, m, hl);
        const v16bf blo = load_b_frag(Blo, ic0 + t * 16, kb, m, hl);
        // S = hi*hi + hi*lo + lo*hi  (~fp32-accurate dot product)
        acc[t] = __builtin_amdgcn_wmma_f32_16x16x32_bf16(false, ahi, false, bhi,
                                                         (short)0, acc[t], false, false);
        acc[t] = __builtin_amdgcn_wmma_f32_16x16x32_bf16(false, ahi, false, blo,
                                                         (short)0, acc[t], false, false);
        acc[t] = __builtin_amdgcn_wmma_f32_16x16x32_bf16(false, alo, false, bhi,
                                                         (short)0, acc[t], false, false);
      }
    }

    // Epilogue: C layout -> VGPR p holds (M = p + hl*8, N = m).
    float na[8], bb[8];
#pragma unroll
    for (int p = 0; p < 8; ++p) {
      const int j = jbase + hl * 8 + p;
      na[p] = anorm[j];
      bb[p] = abias[j];
    }
#pragma unroll
    for (int t = 0; t < 4; ++t) {
      float colsum = 0.f;
#pragma unroll
      for (int p = 0; p < 8; ++p) {
        const float d2 = na[p] + nli[t] - 2.f * acc[t][p];
        const float d  = sqrtf(fmaxf(d2, 0.f)) + EPS_F;
        colsum += __expf(bb[p] - d);
      }
      colsum += __shfl_xor(colsum, 16, 32);       // combine M=0..7 with M=8..15
      if (lane < 16) lds_part[wave][t * 16 + m] = colsum;
    }
    __syncthreads();
    if (threadIdx.x < 64) {
      float s = 0.f;
#pragma unroll
      for (int w = 0; w < 8; ++w) s += lds_part[w][threadIdx.x];
      colacc += s;
    }
    __syncthreads();
  }
  if (threadIdx.x < 64)
    colPart[(size_t)blockIdx.y * colStride + ic0 + threadIdx.x] = colacc;
}

// ---------------------------------------------------------------------------
// Link (edge) term: one wave per edge, each lane covers 4 of the 128 dims.
// Per-block partial sums written without global atomics (deterministic).
// ---------------------------------------------------------------------------
__global__ __launch_bounds__(256) void ldm_edge_kernel(
    const float* __restrict__ L, const float* __restrict__ R, const float* __restrict__ U,
    const float* __restrict__ rho, const float* __restrict__ nu, const float* __restrict__ tau,
    const float* __restrict__ w,
    const int* __restrict__ si, const int* __restrict__ sj, const int* __restrict__ sk,
    int E, float* __restrict__ partials) {
  const int lane = threadIdx.x & 31;
  const int wid  = (blockIdx.x * blockDim.x + threadIdx.x) >> 5;
  const int nw   = (gridDim.x * blockDim.x) >> 5;
  float acc = 0.f;
  for (int e = wid; e < E; e += nw) {
    const int i = si[e], j = sj[e], k = sk[e];
    const v4f lv = *(const v4f*)(L + (size_t)i * 128 + lane * 4);
    const v4f rv = *(const v4f*)(R + (size_t)j * 128 + lane * 4);
    const v4f uv = *(const v4f*)(U + (size_t)k * 128 + lane * 4);
    float slr = 0.f, slu = 0.f;
#pragma unroll
    for (int d = 0; d < 4; ++d) {
      const float dl = lv[d] - rv[d] + EPS_F;
      const float du = lv[d] - uv[d] + EPS_F;
      slr += dl * dl;
      slu += du * du;
    }
#pragma unroll
    for (int off = 16; off > 0; off >>= 1) {
      slr += __shfl_xor(slr, off, 32);
      slu += __shfl_xor(slu, off, 32);
    }
    if (lane == 0)
      acc += w[e] * (rho[i] + nu[j] + tau[k] - sqrtf(slr) - sqrtf(slu));
  }
  __shared__ float sb[8];
  if (lane == 0) sb[threadIdx.x >> 5] = acc;
  __syncthreads();
  if (threadIdx.x == 0) {
    float t = 0.f;
#pragma unroll
    for (int q = 0; q < 8; ++q) t += sb[q];
    partials[blockIdx.x] = t;
  }
}

// ---------------------------------------------------------------------------
// out = sum(edge partials)
//     - sum_i (sum_p colrlPart[p][i]) * exp(rho[i]) * (sum_p coluPart[p][i])
// Single block, fixed-order reduction -> deterministic.
// ---------------------------------------------------------------------------
__global__ __launch_bounds__(256) void ldm_finalize_kernel(
    const float* __restrict__ colrlPart, const float* __restrict__ coluPart,
    const float* __restrict__ rho, int I, int nparts,
    const float* __restrict__ partials, int nb, float* __restrict__ out) {
  __shared__ float sb[256];
  float s = 0.f;
  for (int i = threadIdx.x; i < I; i += 256) {
    float crl = 0.f, cu = 0.f;
    for (int p = 0; p < nparts; ++p) {
      crl += colrlPart[(size_t)p * I + i];
      cu  += coluPart[(size_t)p * I + i];
    }
    s -= crl * __expf(rho[i]) * cu;
  }
  for (int b = threadIdx.x; b < nb; b += 256) s += partials[b];
  sb[threadIdx.x] = s;
  __syncthreads();
  for (int st = 128; st > 0; st >>= 1) {
    if (threadIdx.x < st) sb[threadIdx.x] += sb[threadIdx.x + st];
    __syncthreads();
  }
  if (threadIdx.x == 0) out[0] = sb[0];
}

// ---------------------------------------------------------------------------
extern "C" void kernel_launch(void* const* d_in, const int* in_sizes, int n_in,
                              void* d_out, int out_size, void* d_ws, size_t ws_size,
                              hipStream_t stream) {
  const float* latent_l = (const float*)d_in[0];
  const float* latent_r = (const float*)d_in[1];
  const float* latent_u = (const float*)d_in[2];
  const float* rho      = (const float*)d_in[3];
  const float* nu       = (const float*)d_in[4];
  const float* tau      = (const float*)d_in[5];
  const float* sw       = (const float*)d_in[6];
  const int*   si       = (const int*)d_in[7];
  const int*   sj       = (const int*)d_in[8];
  const int*   sk       = (const int*)d_in[9];

  const int I = in_sizes[3];   // 16384
  const int J = in_sizes[4];   // 8192
  const int K = in_sizes[5];   // 8192
  const int E = in_sizes[6];   // 2000000
  const int NBE = 1024;        // edge partial blocks
  const int JSPLIT = 4;        // J/K partitions of the rate GEMM

  // Workspace carve-out (~17 MB total; all 256B aligned).
  char* ws = (char*)d_ws;
  size_t off = 0;
  auto take = [&](size_t bytes) -> char* {
    char* p = ws + off;
    off += (bytes + 255) & ~(size_t)255;
    return p;
  };
  __bf16* Lhi = (__bf16*)take((size_t)I * 128 * sizeof(__bf16));
  __bf16* Llo = (__bf16*)take((size_t)I * 128 * sizeof(__bf16));
  __bf16* Rhi = (__bf16*)take((size_t)J * 128 * sizeof(__bf16));
  __bf16* Rlo = (__bf16*)take((size_t)J * 128 * sizeof(__bf16));
  __bf16* Uhi = (__bf16*)take((size_t)K * 128 * sizeof(__bf16));
  __bf16* Ulo = (__bf16*)take((size_t)K * 128 * sizeof(__bf16));
  float* nL = (float*)take((size_t)I * sizeof(float));
  float* nR = (float*)take((size_t)J * sizeof(float));
  float* nU = (float*)take((size_t)K * sizeof(float));
  float* colrlPart = (float*)take((size_t)JSPLIT * I * sizeof(float));
  float* coluPart  = (float*)take((size_t)JSPLIT * I * sizeof(float));
  float* partials  = (float*)take((size_t)NBE * sizeof(float));

  // 1) bf16 hi/lo split + norms (L gets no EPS shift; R/U get +EPS).
  ldm_prep_kernel<<<(I + 7) / 8, 256, 0, stream>>>(latent_l, I, 0.f,   Lhi, Llo, nL);
  ldm_prep_kernel<<<(J + 7) / 8, 256, 0, stream>>>(latent_r, J, EPS_F, Rhi, Rlo, nR);
  ldm_prep_kernel<<<(K + 7) / 8, 256, 0, stream>>>(latent_u, K, EPS_F, Uhi, Ulo, nU);

  // 2) Fused WMMA distance/exp column sums for both rate matrices.
  ldm_rate_colsum_kernel<<<dim3(I / 64, JSPLIT), 256, 0, stream>>>(
      Rhi, Rlo, nR, nu, Lhi, Llo, nL, J / JSPLIT, I, colrlPart);
  ldm_rate_colsum_kernel<<<dim3(I / 64, JSPLIT), 256, 0, stream>>>(
      Uhi, Ulo, nU, tau, Lhi, Llo, nL, K / JSPLIT, I, coluPart);

  // 3) Sampled-edge link term (L2-resident gathers).
  ldm_edge_kernel<<<NBE, 256, 0, stream>>>(latent_l, latent_r, latent_u,
                                           rho, nu, tau, sw, si, sj, sk,
                                           E, partials);

  // 4) Final deterministic combine: z_pdist2 - z_pdist1.
  ldm_finalize_kernel<<<1, 256, 0, stream>>>(colrlPart, coluPart, rho, I, JSPLIT,
                                             partials, NBE, (float*)d_out);
}
